// CrystalGraphConvNet_85143431676006
// MI455X (gfx1250) — compile-verified, matrix operand order checked by reading, not run on hardware
//
#include <hip/hip_runtime.h>
#include <math.h>

typedef __attribute__((ext_vector_type(16))) __bf16 v16bf;
typedef __attribute__((ext_vector_type(8)))  __bf16 v8bf;
typedef __attribute__((ext_vector_type(8)))  float  v8f;

#define EPSBN 1e-5f
#define NN 60000
#define MM 12
#define FF 64
#define EE 41
#define HH 128
#define NCC 1500
#define ACNT 40
#define KCONV 169          // 2F + E
#define KPAD_CONV 192
#define TWOF 128
#define ORIG 92
#define KPAD_EMB 96
#define ASTRIDE 216        // conv A-tile LDS stride (16B aligned, bank-spread)

__device__ __forceinline__ unsigned short f2bf(float f) {
  unsigned int u = __float_as_uint(f);
  u += 0x7FFFu + ((u >> 16) & 1u);           // RNE
  return (unsigned short)(u >> 16);
}
__device__ __forceinline__ float softplusf(float x) {
  return log1pf(__expf(-fabsf(x))) + fmaxf(x, 0.0f);
}
__device__ __forceinline__ float sigmoidf_(float x) {
  return 1.0f / (1.0f + __expf(-x));
}
__device__ __forceinline__ v16bf cat16(v8bf lo, v8bf hi) {
  return __builtin_shufflevector(lo, hi, 0,1,2,3,4,5,6,7,8,9,10,11,12,13,14,15);
}

// ---------------- weight prepack: f32 [K,cols] -> bf16 [cols][KPAD] ----------------
__global__ void prepack_fc(const float* __restrict__ W, unsigned short* __restrict__ Wb) {
  int e = blockIdx.x * 256 + threadIdx.x;
  if (e < 3 * TWOF * KPAD_CONV) {
    int l = e / (TWOF * KPAD_CONV);
    int rem = e - l * (TWOF * KPAD_CONV);
    int col = rem / KPAD_CONV, k = rem - col * KPAD_CONV;
    float v = (k < KCONV) ? W[(l * KCONV + k) * TWOF + col] : 0.0f;
    Wb[e] = f2bf(v);
  }
}
__global__ void prepack_mat(const float* __restrict__ W, unsigned short* __restrict__ Wb,
                            int K, int cols, int KPAD) {
  int e = blockIdx.x * 256 + threadIdx.x;
  if (e < cols * KPAD) {
    int col = e / KPAD, k = e - col * KPAD;
    float v = (k < K) ? W[k * cols + col] : 0.0f;
    Wb[e] = f2bf(v);
  }
}

__global__ void zero_f32(float* p, int n) {
  int i = blockIdx.x * 256 + threadIdx.x;
  if (i < n) p[i] = 0.0f;
}

// ---------------- generic WMMA GEMM: C = postop(A @ W + bias), A row-major f32 ----------------
template <int POST>
__global__ __launch_bounds__(256) void gemm_rowmajor(
    const float* __restrict__ A, const unsigned short* __restrict__ Wb,
    const float* __restrict__ bias, float* __restrict__ C,
    int rows, int K, int KPAD, int cols) {
  __shared__ __align__(16) unsigned short Alds[32 * (KPAD_EMB + 24)];
  const int STRIDE = KPAD + 24;                 // 96->120, 64->88 : 16B-aligned rows
  const int tid = threadIdx.x;
  const long long rowBase = (long long)blockIdx.x * 32;
  for (int slot = tid; slot < 32 * KPAD; slot += 256) {
    int row = slot / KPAD, k = slot - row * KPAD;
    long long gr = rowBase + row;
    float v = (gr < rows && k < K) ? A[gr * K + k] : 0.0f;
    Alds[row * STRIDE + k] = f2bf(v);
  }
  __syncthreads();
  const int wave = tid >> 5, lane = tid & 31;
  const int n = lane & 15, khalf = (lane >> 4) << 3;
  const int nct = cols >> 4, nchunks = KPAD >> 5;
  for (int t = wave; t < 2 * nct; t += 8) {
    int rt = t & 1, ct = t >> 1;
    int col = ct * 16 + n;
    float bc = bias[col];
    v8f acc;
#pragma unroll
    for (int i = 0; i < 8; ++i) acc[i] = bc;
    const unsigned short* wcol = Wb + col * KPAD;
    const unsigned short* arow = Alds + (rt * 16 + n) * STRIDE;
    for (int kc = 0; kc < nchunks; ++kc) {
      int k0 = kc * 32 + khalf;
      v8bf alo = *reinterpret_cast<const v8bf*>(arow + k0);
      v8bf ahi = *reinterpret_cast<const v8bf*>(arow + k0 + 16);
      v8bf blo = *reinterpret_cast<const v8bf*>(wcol + k0);
      v8bf bhi = *reinterpret_cast<const v8bf*>(wcol + k0 + 16);
      acc = __builtin_amdgcn_wmma_f32_16x16x32_bf16(false, cat16(alo, ahi),
                                                    false, cat16(blo, bhi),
                                                    (short)0, acc, false, false);
    }
    int rlocal = rt * 16 + ((lane >> 4) << 3);
#pragma unroll
    for (int i = 0; i < 8; ++i) {
      long long r = rowBase + rlocal + i;
      if (r < rows) {
        float v = acc[i];
        if (POST == 1) v = softplusf(v);
        C[r * cols + col] = v;
      }
    }
  }
}

// ---------------- conv pass 1: BN1 column sums / sumsq of gated (GEMM only) ----------------
__global__ __launch_bounds__(256) void conv_pass1(
    const float* __restrict__ x, const float* __restrict__ nbr_fea,
    const int* __restrict__ nbr_idx, const unsigned short* __restrict__ Wb,
    const float* __restrict__ bias, float* __restrict__ stats1) {
  __shared__ __align__(16) unsigned short Alds[48 * ASTRIDE];
  __shared__ int rowSelf[48];
  __shared__ int rowNbr[48];
  __shared__ float lsum[TWOF];
  __shared__ float lsq[TWOF];
  const int tid = threadIdx.x;
  const long long rowBase = (long long)blockIdx.x * 48;
  if (tid < 48) {
    long long gr = rowBase + tid;
    rowSelf[tid] = (int)(gr / MM);
    rowNbr[tid] = nbr_idx[gr];
  }
  if (tid < TWOF) { lsum[tid] = 0.0f; lsq[tid] = 0.0f; }
  __syncthreads();
  for (int slot = tid; slot < 48 * KPAD_CONV; slot += 256) {
    int row = slot / KPAD_CONV, k = slot - row * KPAD_CONV;
    float v;
    if (k < FF)            v = x[(long long)rowSelf[row] * FF + k];
    else if (k < 2 * FF)   v = x[(long long)rowNbr[row] * FF + (k - FF)];
    else if (k < KCONV)    v = nbr_fea[(rowBase + row) * EE + (k - 2 * FF)];
    else                   v = 0.0f;
    Alds[row * ASTRIDE + k] = f2bf(v);
  }
  __syncthreads();
  const int wave = tid >> 5, lane = tid & 31;
  const int n = lane & 15, khalf = (lane >> 4) << 3;
  const int col = wave * 16 + n;
  const float bc = bias[col];
  const unsigned short* wcol = Wb + col * KPAD_CONV;
  __builtin_prefetch(wcol, 0, 0);
  for (int rt = 0; rt < 3; ++rt) {
    v8f acc;
#pragma unroll
    for (int i = 0; i < 8; ++i) acc[i] = bc;
    const unsigned short* arow = Alds + (rt * 16 + n) * ASTRIDE;
#pragma unroll
    for (int kc = 0; kc < 6; ++kc) {
      int k0 = kc * 32 + khalf;
      v8bf alo = *reinterpret_cast<const v8bf*>(arow + k0);
      v8bf ahi = *reinterpret_cast<const v8bf*>(arow + k0 + 16);
      v8bf blo = *reinterpret_cast<const v8bf*>(wcol + k0);
      v8bf bhi = *reinterpret_cast<const v8bf*>(wcol + k0 + 16);
      acc = __builtin_amdgcn_wmma_f32_16x16x32_bf16(false, cat16(alo, ahi),
                                                    false, cat16(blo, bhi),
                                                    (short)0, acc, false, false);
    }
    float s = 0.0f, q = 0.0f;
#pragma unroll
    for (int i = 0; i < 8; ++i) { float v = acc[i]; s += v; q += v * v; }
    atomicAdd(&lsum[col], s);
    atomicAdd(&lsq[col], q);
  }
  __syncthreads();
  if (tid < TWOF)          atomicAdd(&stats1[tid], lsum[tid]);
  else                     atomicAdd(&stats1[tid], lsq[tid - TWOF]);
}

// ---- BN1 finalize: fused scale/shift; also zeroes BN2 stats for this layer ----
__global__ void finalize_bn1(const float* __restrict__ stats1, const float* __restrict__ g,
                             const float* __restrict__ b, float* __restrict__ ss1,
                             float* __restrict__ stats2) {
  int t = threadIdx.x;                         // 128 threads
  const float inv = 1.0f / (float)(NN * MM);
  float mean = stats1[t] * inv;
  float var  = stats1[TWOF + t] * inv - mean * mean;
  float sc = g[t] * rsqrtf(var + EPSBN);
  ss1[t] = sc;
  ss1[TWOF + t] = b[t] - mean * sc;
  stats2[t] = 0.0f;                            // 128 floats = sum[64]+sumsq[64]
}

// ---------------- conv pass 3: recompute GEMM, BN1, gate, neighbor-sum, BN2 stats ----------------
__global__ __launch_bounds__(256) void conv_pass3(
    const float* __restrict__ x, const float* __restrict__ nbr_fea,
    const int* __restrict__ nbr_idx, const unsigned short* __restrict__ Wb,
    const float* __restrict__ bias, const float* __restrict__ ss1,
    float* __restrict__ summed, float* __restrict__ stats2) {
  __shared__ __align__(16) unsigned short Alds[48 * ASTRIDE];
  __shared__ int rowSelf[48];
  __shared__ int rowNbr[48];
  __shared__ float G[48 * TWOF];
  __shared__ float lsum[FF];
  __shared__ float lsq[FF];
  const int tid = threadIdx.x;
  const long long rowBase = (long long)blockIdx.x * 48;   // 4 whole atoms
  if (tid < 48) {
    long long gr = rowBase + tid;
    rowSelf[tid] = (int)(gr / MM);
    rowNbr[tid] = nbr_idx[gr];
  }
  if (tid < FF) { lsum[tid] = 0.0f; lsq[tid] = 0.0f; }
  __syncthreads();
  for (int slot = tid; slot < 48 * KPAD_CONV; slot += 256) {
    int row = slot / KPAD_CONV, k = slot - row * KPAD_CONV;
    float v;
    if (k < FF)            v = x[(long long)rowSelf[row] * FF + k];
    else if (k < 2 * FF)   v = x[(long long)rowNbr[row] * FF + (k - FF)];
    else if (k < KCONV)    v = nbr_fea[(rowBase + row) * EE + (k - 2 * FF)];
    else                   v = 0.0f;
    Alds[row * ASTRIDE + k] = f2bf(v);
  }
  __syncthreads();
  const int wave = tid >> 5, lane = tid & 31;
  const int n = lane & 15, khalf = (lane >> 4) << 3;
  const int col = wave * 16 + n;
  const float bc = bias[col];
  const float sc = ss1[col];
  const float sh = ss1[TWOF + col];
  const unsigned short* wcol = Wb + col * KPAD_CONV;
  __builtin_prefetch(wcol, 0, 0);
  for (int rt = 0; rt < 3; ++rt) {
    v8f acc;
#pragma unroll
    for (int i = 0; i < 8; ++i) acc[i] = bc;
    const unsigned short* arow = Alds + (rt * 16 + n) * ASTRIDE;
#pragma unroll
    for (int kc = 0; kc < 6; ++kc) {
      int k0 = kc * 32 + khalf;
      v8bf alo = *reinterpret_cast<const v8bf*>(arow + k0);
      v8bf ahi = *reinterpret_cast<const v8bf*>(arow + k0 + 16);
      v8bf blo = *reinterpret_cast<const v8bf*>(wcol + k0);
      v8bf bhi = *reinterpret_cast<const v8bf*>(wcol + k0 + 16);
      acc = __builtin_amdgcn_wmma_f32_16x16x32_bf16(false, cat16(alo, ahi),
                                                    false, cat16(blo, bhi),
                                                    (short)0, acc, false, false);
    }
    int rowoff = rt * 16 + ((lane >> 4) << 3);
#pragma unroll
    for (int i = 0; i < 8; ++i)
      G[(rowoff + i) * TWOF + col] = acc[i] * sc + sh;     // BN1 applied
  }
  __syncthreads();
  // gate + reduce over 12 neighbors: thread -> (atom 0..3, feature 0..63)
  int a = tid >> 6, c = tid & 63;
  float s = 0.0f;
#pragma unroll
  for (int r = 0; r < MM; ++r) {
    const float* row = &G[(a * MM + r) * TWOF];
    s += sigmoidf_(row[c]) * softplusf(row[FF + c]);
  }
  long long atom = (long long)blockIdx.x * 4 + a;
  summed[atom * FF + c] = s;
  atomicAdd(&lsum[c], s);
  atomicAdd(&lsq[c], s * s);
  __syncthreads();
  if (tid < FF)            atomicAdd(&stats2[tid], lsum[tid]);
  else if (tid < 2 * FF)   atomicAdd(&stats2[tid], lsq[tid - FF]);
}

__global__ void finalize_bn2(const float* __restrict__ stats2, const float* __restrict__ g,
                             const float* __restrict__ b, float* __restrict__ ss2) {
  int t = threadIdx.x;                         // 64 threads
  const float inv = 1.0f / (float)NN;
  float mean = stats2[t] * inv;
  float var  = stats2[FF + t] * inv - mean * mean;
  float sc = g[t] * rsqrtf(var + EPSBN);
  ss2[t] = sc;
  ss2[FF + t] = b[t] - mean * sc;
}

// ---- x_new = softplus(x + BN2(summed)) ----
__global__ void conv_update(const float* __restrict__ xin, const float* __restrict__ summed,
                            const float* __restrict__ ss2, float* __restrict__ xout, int total) {
  int i = blockIdx.x * 256 + threadIdx.x;
  if (i < total) {
    int c = i & 63;
    xout[i] = softplusf(xin[i] + summed[i] * ss2[c] + ss2[FF + c]);
  }
}

// ---- per-crystal mean pool + softplus ----
__global__ void pool_softplus(const float* __restrict__ x, const int* __restrict__ cidx,
                              float* __restrict__ p) {
  int i = blockIdx.x * 256 + threadIdx.x;
  if (i < NCC * FF) {
    int c = i >> 6, f = i & 63;
    float s = 0.0f;
    for (int a = 0; a < ACNT; ++a)
      s += x[(long long)cidx[c * ACNT + a] * FF + f];
    p[i] = softplusf(s * (1.0f / (float)ACNT));
  }
}

// ---- final 128 -> 1 projection ----
__global__ void out_proj(const float* __restrict__ h, const float* __restrict__ W,
                         const float* __restrict__ b, float* __restrict__ out) {
  int r = blockIdx.x * 256 + threadIdx.x;
  if (r < NCC) {
    float s = b[0];
#pragma unroll 4
    for (int j = 0; j < HH; ++j) s += h[r * HH + j] * W[j];
    out[r] = s;
  }
}

extern "C" void kernel_launch(void* const* d_in, const int* in_sizes, int n_in,
                              void* d_out, int out_size, void* d_ws, size_t ws_size,
                              hipStream_t stream) {
  (void)in_sizes; (void)n_in; (void)out_size; (void)ws_size;
  const float* atom_fea = (const float*)d_in[0];
  const float* nbr_fea  = (const float*)d_in[1];
  const int*   nbr_idx  = (const int*)d_in[2];
  const int*   cidx     = (const int*)d_in[3];
  const float* emb_W = (const float*)d_in[4];
  const float* emb_b = (const float*)d_in[5];
  const float* fc_W  = (const float*)d_in[6];
  const float* fc_b  = (const float*)d_in[7];
  const float* bn1_g = (const float*)d_in[8];
  const float* bn1_b = (const float*)d_in[9];
  const float* bn2_g = (const float*)d_in[10];
  const float* bn2_b = (const float*)d_in[11];
  const float* fc1_W = (const float*)d_in[12];
  const float* fc1_b = (const float*)d_in[13];
  const float* out_W = (const float*)d_in[14];
  const float* out_b = (const float*)d_in[15];

  float* ws = (float*)d_ws;
  float* x0     = ws;                          // [60000,64]
  float* x1     = ws + 3840000;                // [60000,64]
  float* summed = ws + 7680000;                // [60000,64]
  float* pbuf   = ws + 11520000;               // [1500,64]
  float* hbuf   = ws + 11616000;               // [1500,128]
  float* stats1 = ws + 11808000;               // 256: sum[128]+sumsq[128]
  float* ss1    = stats1 + 256;                // 256: scale[128]+shift[128]
  float* stats2 = ss1 + 256;                   // 128
  float* ss2    = stats2 + 128;                // 128
  unsigned short* wbfc  = (unsigned short*)(ws + 11808768);   // 3*128*192 bf16
  unsigned short* wbemb = wbfc + 3 * TWOF * KPAD_CONV;        // 64*96
  unsigned short* wbfc1 = wbemb + FF * KPAD_EMB;              // 128*64

  // weight prepack (bf16, K-padded, col-major)
  int nfc = 3 * TWOF * KPAD_CONV;
  prepack_fc<<<(nfc + 255) / 256, 256, 0, stream>>>(fc_W, wbfc);
  prepack_mat<<<(FF * KPAD_EMB + 255) / 256, 256, 0, stream>>>(emb_W, wbemb, ORIG, FF, KPAD_EMB);
  prepack_mat<<<(TWOF * FF + 255) / 256, 256, 0, stream>>>(fc1_W, wbfc1, FF, TWOF, FF);

  // embedding: x0 = atom_fea @ emb_W + emb_b
  gemm_rowmajor<0><<<(NN + 31) / 32, 256, 0, stream>>>(atom_fea, wbemb, emb_b, x0,
                                                       NN, ORIG, KPAD_EMB, FF);
  float* xin = x0;
  float* xout = x1;
  for (int l = 0; l < 3; ++l) {
    const unsigned short* wl = wbfc + l * TWOF * KPAD_CONV;
    const float* bl = fc_b + l * TWOF;
    zero_f32<<<1, 256, 0, stream>>>(stats1, 256);
    conv_pass1<<<15000, 256, 0, stream>>>(xin, nbr_fea, nbr_idx, wl, bl, stats1);
    finalize_bn1<<<1, 128, 0, stream>>>(stats1, bn1_g + l * TWOF, bn1_b + l * TWOF, ss1, stats2);
    conv_pass3<<<15000, 256, 0, stream>>>(xin, nbr_fea, nbr_idx, wl, bl, ss1, summed, stats2);
    finalize_bn2<<<1, 64, 0, stream>>>(stats2, bn2_g + l * FF, bn2_b + l * FF, ss2);
    conv_update<<<(NN * FF + 255) / 256, 256, 0, stream>>>(xin, summed, ss2, xout, NN * FF);
    float* t = xin; xin = xout; xout = t;
  }
  pool_softplus<<<(NCC * FF + 255) / 256, 256, 0, stream>>>(xin, cidx, pbuf);
  gemm_rowmajor<1><<<(NCC + 31) / 32, 256, 0, stream>>>(pbuf, wbfc1, fc1_b, hbuf,
                                                        NCC, FF, FF, TWOF);
  out_proj<<<(NCC + 255) / 256, 256, 0, stream>>>(hbuf, out_W, out_b, (float*)d_out);
}
